// CustomAttnProcessor_75737453297854
// MI455X (gfx1250) — matrix operand order repaired and checked by reading, not compiled, not run on hardware
//
#include <hip/hip_runtime.h>

// ---------------------------------------------------------------------------
// MI455X (gfx1250) self-attention. B=8 S=1024 C=1280 H=20 d=64.
// ~150 GFLOP vs ~110MB traffic => compute bound => everything on the bf16 WMMA
// pipe (v_wmma_f32_16x16x32_bf16, 8x MACs/issue vs f32 16x16x4), fp32 accum.
// Data movement uses the CDNA5 async path: global_load_async_to_lds_b128 with
// double buffering + s_wait_asynccnt, fragments read as aligned b128 from LDS.
// ---------------------------------------------------------------------------

typedef __attribute__((ext_vector_type(16))) __bf16   v16bf;
typedef __attribute__((ext_vector_type(8)))  float    v8f;
typedef __attribute__((ext_vector_type(8)))  unsigned v8u;

#define BB 8
#define SS 1024
#define CC 1280
#define HH 20
#define DD 64
#define MM (BB * SS)   // 8192 rows

// fp32 -> bf16 round-to-nearest-even, bit level.
__device__ __forceinline__ unsigned short f2bf(float f) {
  unsigned x = __float_as_uint(f);
  return (unsigned short)((x + 0x7FFFu + ((x >> 16) & 1u)) >> 16);
}
__device__ __forceinline__ unsigned pack2(float lo, float hi) {
  return (unsigned)f2bf(lo) | ((unsigned)f2bf(hi) << 16);
}
// Both WMMA A (16x32) and B (32x16) bf16 fragments are, per lane, two aligned
// 16-byte chunks -> assemble from two uint4 loads.
__device__ __forceinline__ v16bf frag(uint4 lo, uint4 hi) {
  v8u u;
  u[0] = lo.x; u[1] = lo.y; u[2] = lo.z; u[3] = lo.w;
  u[4] = hi.x; u[5] = hi.y; u[6] = hi.z; u[7] = hi.w;
  return __builtin_bit_cast(v16bf, u);
}
// LDS byte offset of a __shared__ object (generic low 32 bits == LDS offset).
__device__ __forceinline__ unsigned lds_off(const void* p) {
  return (unsigned)(size_t)p;
}
__device__ __forceinline__ void async_b128(unsigned lds, const void* gaddr) {
  asm volatile("global_load_async_to_lds_b128 %0, %1, off"
               :: "v"(lds), "v"((unsigned long long)gaddr) : "memory");
}

// ---------------------------------------------------------------------------
__global__ void f32_to_bf16_kernel(const float* __restrict__ in,
                                   uint4* __restrict__ out, int n8) {
  int i = blockIdx.x * blockDim.x + threadIdx.x;
  if (i >= n8) return;
  const float4 a = ((const float4*)in)[2 * i];
  const float4 b = ((const float4*)in)[2 * i + 1];
  uint4 o;
  o.x = pack2(a.x, a.y); o.y = pack2(a.z, a.w);
  o.z = pack2(b.x, b.y); o.w = pack2(b.z, b.w);
  out[i] = o;
}

// ---------------------------------------------------------------------------
// C[m,n] = sum_k A[m,k] * W[n,k]  (A: MMxCC bf16, W: CCxCC bf16, both row-major)
// Workgroup = 8 waves = 64(M) x 256(N); wave = 16 x 128 (8 f32 accum frags).
// B tile (256 cols x 32 K = 16KB) staged in LDS via async b128, double-buffered;
// shared by the 4 waves that need each half. A frags load direct (2 b128/lane).
// mode 0: bf16 head-split [B,H,S,d] (Q,K)   mode 1: bf16 [B,H,d,S] (V)
// mode 2: fp32 [m,n] + bias (output projection)
__global__ void __launch_bounds__(256) proj_wmma_kernel(
    const unsigned short* __restrict__ A, const unsigned short* __restrict__ W,
    const float* __restrict__ bias, unsigned short* __restrict__ outb,
    float* __restrict__ outf, int mode) {
  __shared__ unsigned short Bsh[2][256 * 32];   // [buf][col][k] 32KB

  const int tid  = threadIdx.x;
  const int wave = tid >> 5;
  const int lane = tid & 31;
  const int h16  = lane >> 4;
  const int l15  = lane & 15;
  const int m0 = blockIdx.x * 64 + (wave >> 1) * 16;
  const int gn0 = blockIdx.y * 256;
  const int n0w = (wave & 1) * 128;

  const unsigned ldsB = lds_off(&Bsh[0][0]);
  // Each thread copies one B column (64B) per tile as 4 async b128s.
  const unsigned short* wcol = W + (size_t)(gn0 + tid) * CC;

  // A fragment base: per lane two b128s at elements kk+8*h16 and kk+16+8*h16.
  const uint4* ap = (const uint4*)(A + (size_t)(m0 + l15) * CC + 8 * h16);

  v8f acc[8] = {};

  // prologue: stage tile kk=0 into buf 0
#pragma unroll
  for (int s = 0; s < 4; ++s)
    async_b128(ldsB + tid * 64 + s * 16, wcol + s * 8);

  for (int it = 0; it < CC / 32; ++it) {
    const int kk  = it * 32;
    const int buf = it & 1;
    if (it + 1 < CC / 32) {
#pragma unroll
      for (int s = 0; s < 4; ++s)
        async_b128(ldsB + (buf ^ 1) * 16384 + tid * 64 + s * 16,
                   wcol + kk + 32 + s * 8);
      asm volatile("s_wait_asynccnt 0x4" ::: "memory");
    } else {
      asm volatile("s_wait_asynccnt 0x0" ::: "memory");
    }
    __syncthreads();

    const uint4 alo = ap[kk >> 3];
    const uint4 ahi = ap[(kk >> 3) + 2];
    const v16bf af = frag(alo, ahi);
#pragma unroll
    for (int t = 0; t < 8; ++t) {
      const int col = n0w + t * 16 + l15;     // B-frag column = row of W
      const uint4* bp = (const uint4*)&Bsh[buf][col * 32 + 8 * h16];
      acc[t] = __builtin_amdgcn_wmma_f32_16x16x32_bf16(
          false, af, false, frag(bp[0], bp[2]), (short)0, acc[t], false, false);
    }
    __syncthreads();   // protect buf before it is re-filled
  }

#pragma unroll
  for (int t = 0; t < 8; ++t) {
    const int n  = gn0 + n0w + t * 16 + l15;
    const int h  = n >> 6;
    const int dd = n & 63;
#pragma unroll
    for (int j = 0; j < 8; ++j) {
      const int m = m0 + j + 8 * h16;         // D layout: VGPR j -> row j+8*h16
      const float val = acc[t][j];
      if (mode == 2) {
        outf[(size_t)m * CC + n] = val + bias[n];
      } else {
        const int b = m >> 10, s = m & 1023;
        size_t addr = (mode == 0)
            ? (size_t)(((b * HH + h) * SS + s) * DD + dd)
            : (size_t)(((b * HH + h) * DD + dd) * SS + s);
        outb[addr] = f2bf(val);
      }
    }
  }
}

// ---------------------------------------------------------------------------
// Flash attention for one (b,h) per workgroup: Q,K in [S,64] bf16, V transposed
// [64,S] bf16. 8 waves x 16 query rows = 128 rows per block. K/V 32-key tiles
// staged in LDS via async b128 (shared by all 8 waves), double-buffered.
// Online softmax row stats in registers; row reductions via shfl_xor over the
// 16-lane halves of the C/D layout; P goes through a 1KB/wave LDS tile for
// D-layout -> A-layout conversion.
__global__ void __launch_bounds__(256) attn_wmma_kernel(
    const unsigned short* __restrict__ Qb, const unsigned short* __restrict__ Kb,
    const unsigned short* __restrict__ Vtb, unsigned short* __restrict__ Ctx) {
  __shared__ unsigned short Ksh[2][32 * 64];   // [buf][krow][d]   8KB
  __shared__ unsigned short Vsh[2][64 * 32];   // [buf][dv][k]     8KB
  __shared__ unsigned short Psh[8][16 * 32];   // per-wave P tile  8KB

  const int tid  = threadIdx.x;
  const int wave = tid >> 5;
  const int lane = tid & 31;
  const int h16  = lane >> 4;
  const int l15  = lane & 15;
  const int bh   = blockIdx.x;                 // 0..159
  const int qrow = blockIdx.y * 128 + wave * 16;

  const unsigned short* Qp = Qb  + (size_t)bh * (SS * DD);
  const unsigned short* Kp = Kb  + (size_t)bh * (SS * DD);
  const unsigned short* Vp = Vtb + (size_t)bh * (SS * DD);

  const unsigned ldsK = lds_off(&Ksh[0][0]);
  const unsigned ldsV = lds_off(&Vsh[0][0]);

  // Async-copy assignments: K tile row r=tid>>3, 16B chunk c=tid&7;
  //                         V tile row dv=tid>>2, 16B chunk c=tid&3.
  const int kr = tid >> 3, kc = tid & 7;
  const int vr = tid >> 2, vc = tid & 3;

  // Q A-fragments (GEMM-K = d = 64): d 0..31 and 32..63, 2 b128 each.
  const uint4* qp = (const uint4*)(Qp + (qrow + l15) * DD + 8 * h16);
  const v16bf qa0 = frag(qp[0], qp[2]);
  const v16bf qa1 = frag(qp[4], qp[6]);

  v8f acc[4] = {};
  float mrow[8], lrow[8];
#pragma unroll
  for (int j = 0; j < 8; ++j) { mrow[j] = -3.0e38f; lrow[j] = 0.f; }
  const float scale = 0.125f;  // 64^-0.5

  // prologue: stage K/V tile kb=0 into buf 0
  async_b128(ldsK + kr * 128 + kc * 16, Kp + kr * DD + kc * 8);
  async_b128(ldsV + vr * 64 + vc * 16, Vp + vr * SS + vc * 8);

  for (int it = 0; it < SS / 32; ++it) {
    const int kb  = it * 32;
    const int buf = it & 1;
    if (it + 1 < SS / 32) {
      async_b128(ldsK + (buf ^ 1) * 8192 + kr * 128 + kc * 16,
                 Kp + (kb + 32 + kr) * DD + kc * 8);
      async_b128(ldsV + (buf ^ 1) * 8192 + vr * 64 + vc * 16,
                 Vp + vr * SS + kb + 32 + vc * 8);
      asm volatile("s_wait_asynccnt 0x2" ::: "memory");
    } else {
      asm volatile("s_wait_asynccnt 0x0" ::: "memory");
    }
    __syncthreads();

    // ---- scores: 2 sub-blocks of 16 keys, 2 WMMAs each -------------------
    float t0[8], t1[8];
#pragma unroll
    for (int sub = 0; sub < 2; ++sub) {
      const int krl = sub * 16 + l15;          // B-frag column = key row
      const uint4* kp0 = (const uint4*)&Ksh[buf][krl * DD + 8 * h16];
      const uint4* kp1 = (const uint4*)&Ksh[buf][krl * DD + 32 + 8 * h16];
      v8f sc = {};
      sc = __builtin_amdgcn_wmma_f32_16x16x32_bf16(
          false, qa0, false, frag(kp0[0], kp0[2]), (short)0, sc, false, false);
      sc = __builtin_amdgcn_wmma_f32_16x16x32_bf16(
          false, qa1, false, frag(kp1[0], kp1[2]), (short)0, sc, false, false);
      float* tt = sub ? t1 : t0;
#pragma unroll
      for (int j = 0; j < 8; ++j) tt[j] = sc[j] * scale;
    }

    // Hoist V fragment loads so DS traffic overlaps the softmax VALU work.
    uint4 vlo[4], vhi[4];
#pragma unroll
    for (int t = 0; t < 4; ++t) {
      const uint4* vp = (const uint4*)&Vsh[buf][(t * 16 + l15) * 32 + 8 * h16];
      vlo[t] = vp[0]; vhi[t] = vp[2];
    }

    // ---- online softmax (row j + 8*h16 spans the 16 lanes of a half) -----
#pragma unroll
    for (int j = 0; j < 8; ++j) {
      float bm = fmaxf(t0[j], t1[j]);
      bm = fmaxf(bm, __shfl_xor(bm, 1));
      bm = fmaxf(bm, __shfl_xor(bm, 2));
      bm = fmaxf(bm, __shfl_xor(bm, 4));
      bm = fmaxf(bm, __shfl_xor(bm, 8));
      const float mnew  = fmaxf(mrow[j], bm);
      const float alpha = __expf(mrow[j] - mnew);
      mrow[j] = mnew;
      const float p0 = __expf(t0[j] - mnew);
      const float p1 = __expf(t1[j] - mnew);
      float rs = p0 + p1;
      rs += __shfl_xor(rs, 1);
      rs += __shfl_xor(rs, 2);
      rs += __shfl_xor(rs, 4);
      rs += __shfl_xor(rs, 8);
      lrow[j] = lrow[j] * alpha + rs;
#pragma unroll
      for (int t = 0; t < 4; ++t) acc[t][j] *= alpha;
      const int prow = j + 8 * h16;
      Psh[wave][prow * 32 + l15]      = f2bf(p0);
      Psh[wave][prow * 32 + 16 + l15] = f2bf(p1);
    }
    asm volatile("s_wait_dscnt 0x0" ::: "memory");  // P visible to this wave

    // P as A-fragment (16x32), then 4 PV WMMAs.
    const uint4* pp = (const uint4*)&Psh[wave][l15 * 32 + 8 * h16];
    const v16bf pa = frag(pp[0], pp[2]);
#pragma unroll
    for (int t = 0; t < 4; ++t)
      acc[t] = __builtin_amdgcn_wmma_f32_16x16x32_bf16(
          false, pa, false, frag(vlo[t], vhi[t]), (short)0, acc[t], false, false);

    __syncthreads();   // protect K/V buf before refill
  }

  const int b = bh / HH, h = bh % HH;
#pragma unroll
  for (int j = 0; j < 8; ++j) {
    const float inv = 1.0f / lrow[j];
    const int srow = qrow + j + 8 * h16;
#pragma unroll
    for (int t = 0; t < 4; ++t) {
      const int c = h * DD + t * 16 + l15;
      Ctx[(size_t)(b * SS + srow) * CC + c] = f2bf(acc[t][j] * inv);
    }
  }
}

// ---------------------------------------------------------------------------
extern "C" void kernel_launch(void* const* d_in, const int* in_sizes, int n_in,
                              void* d_out, int out_size, void* d_ws,
                              size_t ws_size, hipStream_t stream) {
  (void)in_sizes; (void)n_in; (void)out_size; (void)ws_size;
  const float* hidden = (const float*)d_in[0];
  const float* Wq = (const float*)d_in[1];
  const float* Wk = (const float*)d_in[2];
  const float* Wv = (const float*)d_in[3];
  const float* Wo = (const float*)d_in[4];
  const float* bo = (const float*)d_in[5];
  float* out = (float*)d_out;

  // Workspace layout (bf16), total ~118 MB.
  unsigned short* ws = (unsigned short*)d_ws;
  size_t off = 0;
  unsigned short* hb  = ws + off; off += (size_t)MM * CC;
  unsigned short* wqb = ws + off; off += (size_t)CC * CC;
  unsigned short* wkb = ws + off; off += (size_t)CC * CC;
  unsigned short* wvb = ws + off; off += (size_t)CC * CC;
  unsigned short* wob = ws + off; off += (size_t)CC * CC;
  unsigned short* Qb  = ws + off; off += (size_t)MM * CC;
  unsigned short* Kb  = ws + off; off += (size_t)MM * CC;
  unsigned short* Vtb = ws + off; off += (size_t)MM * CC;
  unsigned short* ctx = ws + off; off += (size_t)MM * CC;

  auto cvt = [&](const float* src, unsigned short* dst, size_t n) {
    int n8 = (int)(n / 8);
    f32_to_bf16_kernel<<<dim3((n8 + 255) / 256), dim3(256), 0, stream>>>(
        src, (uint4*)dst, n8);
  };
  cvt(hidden, hb, (size_t)MM * CC);
  cvt(Wq, wqb, (size_t)CC * CC);
  cvt(Wk, wkb, (size_t)CC * CC);
  cvt(Wv, wvb, (size_t)CC * CC);
  cvt(Wo, wob, (size_t)CC * CC);

  const dim3 gGemm(MM / 64, CC / 256);   // 128 x 5
  proj_wmma_kernel<<<gGemm, dim3(256), 0, stream>>>(hb, wqb, nullptr, Qb,
                                                    nullptr, 0);
  proj_wmma_kernel<<<gGemm, dim3(256), 0, stream>>>(hb, wkb, nullptr, Kb,
                                                    nullptr, 0);
  proj_wmma_kernel<<<gGemm, dim3(256), 0, stream>>>(hb, wvb, nullptr, Vtb,
                                                    nullptr, 1);

  attn_wmma_kernel<<<dim3(BB * HH, SS / 128), dim3(256), 0, stream>>>(Qb, Kb,
                                                                      Vtb, ctx);

  proj_wmma_kernel<<<gGemm, dim3(256), 0, stream>>>(ctx, wob, bo, nullptr, out,
                                                    2);
}
